// MultiHeadAttention_5523327942876
// MI455X (gfx1250) — matrix operand order repaired and checked by reading, not compiled, not run on hardware
//
#include <hip/hip_runtime.h>

// ---------------------------------------------------------------------------
// Multi-head attention forward for MI455X (gfx1250, wave32, WMMA).
// B=2, S=2048, D=1024, H=16, DH=64.
// d_out = [ out fp32 (B*S*D) | attn fp32 (B*H*S*S) ]
// d_ws  = f16 Qh[B,H,S,64] | f16 Kh[B,H,S,64] | f16 Vt[B,H,64,S] | f16 ctx[B,S,D]
// ---------------------------------------------------------------------------

typedef _Float16 h16;
typedef __attribute__((ext_vector_type(4)))  float v4f;
typedef __attribute__((ext_vector_type(8)))  float v8f;
typedef __attribute__((ext_vector_type(8)))  h16   v8h;
typedef __attribute__((ext_vector_type(16))) h16   v16h;

#define DEV __device__ __forceinline__

constexpr int BB = 2, SS = 2048, DD = 1024, HH = 16, DHH = 64;
constexpr int LSTR = 48;              // LDS tile row stride in halves (96 B)
constexpr float NEG_INF = -__builtin_inff();

DEV v16h hcat(v8h lo, v8h hi) {
  return __builtin_shufflevector(lo, hi, 0,1,2,3,4,5,6,7,8,9,10,11,12,13,14,15);
}

// A-fragment (16x32 f16) from an LDS tile. Lane L -> row (L&15),
// 16 contiguous K starting at (L>>4)*16.
DEV v16h a_frag_lds(const h16* tile, int row_off) {
  const int lane = threadIdx.x & 31;
  const h16* p = tile + (row_off + (lane & 15)) * LSTR + ((lane >> 4) * 16);
  return hcat(*(const v8h*)p, *(const v8h*)(p + 8));
}

// B-fragment (32x16 f16) from an LDS tile. Lane L -> column N=(L&15),
// 16 contiguous K starting at (L>>4)*16.
DEV v16h b_frag_lds(const h16* tile, int row_off) {
  const int lane = threadIdx.x & 31;
  return *(const v16h*)(tile + (row_off + (lane & 15)) * LSTR + ((lane >> 4) * 16));
}

// Stage a 128x32 tile: fp32 global -> f16 LDS (256 threads, 16 elems each).
DEV void stage_f32(h16* dst, const float* src, int ld, int row0, int k0) {
  const int t = threadIdx.x;
  const int r = t >> 1, kc = (t & 1) * 16;
  const float* s = src + (size_t)(row0 + r) * ld + k0 + kc;
  v4f x0 = *(const v4f*)(s + 0),  x1 = *(const v4f*)(s + 4);
  v4f x2 = *(const v4f*)(s + 8),  x3 = *(const v4f*)(s + 12);
  v8h o0, o1;
#pragma unroll
  for (int i = 0; i < 4; ++i) {
    o0[i] = (h16)x0[i]; o0[4 + i] = (h16)x1[i];
    o1[i] = (h16)x2[i]; o1[4 + i] = (h16)x3[i];
  }
  *(v8h*)(dst + r * LSTR + kc)     = o0;
  *(v8h*)(dst + r * LSTR + kc + 8) = o1;
}

// Stage a 128x32 tile: f16 global -> f16 LDS.
DEV void stage_f16(h16* dst, const h16* src, int ld, int row0, int k0) {
  const int t = threadIdx.x;
  const int r = t >> 1, kc = (t & 1) * 16;
  const h16* s = src + (size_t)(row0 + r) * ld + k0 + kc;
  *(v8h*)(dst + r * LSTR + kc)     = *(const v8h*)s;
  *(v8h*)(dst + r * LSTR + kc + 8) = *(const v8h*)(s + 8);
}

// ---------------------------------------------------------------------------
// NT GEMM: C[m,n] = sum_k A[m,k] * W[n,k] + bias[n], M=4096, N=K=1024.
// MODE 0: out f16, head layout [B,H,S,64] (Q/K projections, scale folded in)
// MODE 1: out f16, transposed head layout [B,H,64,S] (V projection)
// MODE 2: out fp32 row-major [M,N] (final projection, A is f16 context)
// Block = 256 threads (8 waves), tile 128x128, k-step 32.
// ---------------------------------------------------------------------------
template <int MODE, bool AHALF>
__global__ __launch_bounds__(256) void gemm_nt_kernel(const void* __restrict__ A,
                                                      const float* __restrict__ W,
                                                      const float* __restrict__ bias,
                                                      void* __restrict__ Out,
                                                      float scale) {
  constexpr int N = DD, K = DD;
  __shared__ h16 At[128 * LSTR];
  __shared__ h16 Bt[128 * LSTR];

  const int m0 = blockIdx.y * 128;
  const int n0 = blockIdx.x * 128;
  const int wid = threadIdx.x >> 5;
  const int wm = wid >> 1;          // 0..3 : 32-row strip
  const int wn = wid & 1;           // 0..1 : 64-col strip
  const int lane = threadIdx.x & 31;

  v8f acc[2][4] = {};

  for (int k0 = 0; k0 < K; k0 += 32) {
    __syncthreads();
    if constexpr (AHALF) stage_f16(At, (const h16*)A, K, m0, k0);
    else                 stage_f32(At, (const float*)A, K, m0, k0);
    stage_f32(Bt, W, K, n0, k0);
    __syncthreads();

    v16h af[2], bf[4];
#pragma unroll
    for (int fm = 0; fm < 2; ++fm) af[fm] = a_frag_lds(At, wm * 32 + fm * 16);
#pragma unroll
    for (int fn = 0; fn < 4; ++fn) bf[fn] = b_frag_lds(Bt, wn * 64 + fn * 16);
#pragma unroll
    for (int fm = 0; fm < 2; ++fm)
#pragma unroll
      for (int fn = 0; fn < 4; ++fn)
        acc[fm][fn] = __builtin_amdgcn_wmma_f32_16x16x32_f16(
            false, af[fm], false, bf[fn], (short)0, acc[fm][fn], false, false);
  }

  // Epilogue. C layout: VGPR r -> M = (lane>=16 ? 8 : 0) + r, N = lane&15.
#pragma unroll
  for (int fm = 0; fm < 2; ++fm) {
#pragma unroll
    for (int fn = 0; fn < 4; ++fn) {
      const int ng = n0 + wn * 64 + fn * 16 + (lane & 15);
      const float bv = bias[ng];
#pragma unroll
      for (int r = 0; r < 8; ++r) {
        const int mg = m0 + wm * 32 + fm * 16 + ((lane >> 4) * 8) + r;
        const float val = (acc[fm][fn][r] + bv) * scale;
        if constexpr (MODE == 0) {
          const int bi = mg / SS, si = mg % SS;
          const int hi = ng / DHH, di = ng % DHH;
          ((h16*)Out)[(((size_t)(bi * HH + hi) * SS + si) * DHH) + di] = (h16)val;
        } else if constexpr (MODE == 1) {
          const int bi = mg / SS, si = mg % SS;
          const int hi = ng / DHH, di = ng % DHH;
          ((h16*)Out)[(((size_t)(bi * HH + hi) * DHH + di) * SS) + si] = (h16)val;
        } else {
          ((float*)Out)[(size_t)mg * N + ng] = val;
        }
      }
    }
  }
}

// ---------------------------------------------------------------------------
// Scores + causal softmax. One block = 16 query rows of one (b,h).
// 8 waves sweep the 2048 key columns (skipping fully-masked chunks),
// raw fp32 scores land in 128 KB dynamic LDS, then a row-parallel softmax
// writes normalized probabilities (0 in masked region) to the attn output.
// 1/sqrt(DH) is pre-folded into Qh.
// ---------------------------------------------------------------------------
__global__ __launch_bounds__(256) void attn_scores_kernel(const h16* __restrict__ Qh,
                                                          const h16* __restrict__ Kh,
                                                          float* __restrict__ attn) {
  extern __shared__ float sc[];                 // [16][SS]
  const int qb = blockIdx.x % (SS / 16);
  const int bh = blockIdx.x / (SS / 16);
  const int qbase = qb * 16;
  const int wn = threadIdx.x >> 5;
  const int lane = threadIdx.x & 31;

  const h16* Qb = Qh + (size_t)bh * SS * DHH;
  const h16* Kb = Kh + (size_t)bh * SS * DHH;

  // Hoist Q fragments for K-steps 0 and 32 (DH = 64).
  v16h aq[2];
  {
    const int m = qbase + (lane & 15);
    const int kb = (lane >> 4) * 16;
    const h16* p = Qb + (size_t)m * DHH + kb;
    aq[0] = hcat(*(const v8h*)(p + 0),  *(const v8h*)(p + 8));
    aq[1] = hcat(*(const v8h*)(p + 32), *(const v8h*)(p + 40));
  }

  for (int col0 = wn * 16; col0 <= qbase + 15; col0 += 128) {
    v8f c = {};
#pragma unroll
    for (int ks = 0; ks < 2; ++ks) {
      const int n = col0 + (lane & 15);
      const h16* p = Kb + (size_t)n * DHH + ks * 32 + ((lane >> 4) * 16);
      v16h bf = *(const v16h*)p;
      c = __builtin_amdgcn_wmma_f32_16x16x32_f16(false, aq[ks], false, bf,
                                                 (short)0, c, false, false);
    }
    const int col = col0 + (lane & 15);
#pragma unroll
    for (int r = 0; r < 8; ++r) {
      const int mrow = ((lane >> 4) * 8) + r;
      const int q = qbase + mrow;
      sc[mrow * SS + col] = (col <= q) ? c[r] : NEG_INF;
    }
  }
  __syncthreads();

  // Row softmax: 16 threads per row.
  const int row = threadIdx.x >> 4;
  const int tp = threadIdx.x & 15;
  const int q = qbase + row;
  float* srow = sc + row * SS;

  float mx = NEG_INF;
  for (int col = tp; col <= q; col += 16) mx = fmaxf(mx, srow[col]);
#pragma unroll
  for (int off = 8; off; off >>= 1) mx = fmaxf(mx, __shfl_xor(mx, off));

  float sum = 0.f;
  for (int col = tp; col <= q; col += 16) {
    const float e = __expf(srow[col] - mx);
    srow[col] = e;
    sum += e;
  }
#pragma unroll
  for (int off = 8; off; off >>= 1) sum += __shfl_xor(sum, off);
  const float inv = 1.0f / sum;

  float* arow = attn + ((size_t)bh * SS + q) * SS;
  for (int col = tp; col < SS; col += 16)
    arow[col] = (col <= q) ? srow[col] * inv : 0.0f;
}

// ---------------------------------------------------------------------------
// PV: ctx[b, s, h*64+d] = sum_k P[q,k] * Vt[d,k].  NT via transposed V.
// Block = 256 threads, tile 128(q) x 64(d); K-loop capped at causal bound.
// ---------------------------------------------------------------------------
__global__ __launch_bounds__(256) void attn_pv_kernel(const float* __restrict__ attn,
                                                      const h16* __restrict__ Vt,
                                                      h16* __restrict__ ctx) {
  __shared__ h16 Pt[128 * LSTR];
  const int mt = blockIdx.x % (SS / 128);
  const int bh = blockIdx.x / (SS / 128);
  const int bi = bh / HH, hi = bh % HH;
  const int qbase = mt * 128;
  const int wid = threadIdx.x >> 5;
  const int lane = threadIdx.x & 31;

  const float* Ab = attn + (size_t)bh * SS * SS;
  const h16* Vb = Vt + (size_t)bh * DHH * SS;

  v8f acc[4] = {};
  const int kend = qbase + 128;                 // probs are 0 beyond the diagonal
  for (int k0 = 0; k0 < kend; k0 += 32) {
    __syncthreads();
    stage_f32(Pt, Ab, SS, qbase, k0);
    __syncthreads();

    const v16h af = a_frag_lds(Pt, wid * 16);
#pragma unroll
    for (int fn = 0; fn < 4; ++fn) {
      const int n = fn * 16 + (lane & 15);
      const h16* p = Vb + (size_t)n * SS + k0 + ((lane >> 4) * 16);
      v16h bf = *(const v16h*)p;
      acc[fn] = __builtin_amdgcn_wmma_f32_16x16x32_f16(false, af, false, bf,
                                                       (short)0, acc[fn], false, false);
    }
  }

#pragma unroll
  for (int fn = 0; fn < 4; ++fn) {
    const int di = fn * 16 + (lane & 15);
#pragma unroll
    for (int r = 0; r < 8; ++r) {
      const int sq = qbase + wid * 16 + ((lane >> 4) * 8) + r;
      ctx[((size_t)(bi * SS + sq)) * DD + hi * DHH + di] = (h16)acc[fn][r];
    }
  }
}

// ---------------------------------------------------------------------------
extern "C" void kernel_launch(void* const* d_in, const int* in_sizes, int n_in,
                              void* d_out, int out_size, void* d_ws, size_t ws_size,
                              hipStream_t stream) {
  (void)in_sizes; (void)n_in; (void)out_size; (void)ws_size;
  const float* q  = (const float*)d_in[0];
  const float* k  = (const float*)d_in[1];
  const float* v  = (const float*)d_in[2];
  // d_in[3] = mask (static lower-triangular; causality implemented directly)
  const float* Wq = (const float*)d_in[4];
  const float* bq = (const float*)d_in[5];
  const float* Wk = (const float*)d_in[6];
  const float* bk = (const float*)d_in[7];
  const float* Wv = (const float*)d_in[8];
  const float* bv = (const float*)d_in[9];
  const float* Wo = (const float*)d_in[10];
  const float* bo = (const float*)d_in[11];

  float* out  = (float*)d_out;
  float* attn = out + (size_t)BB * SS * DD;

  const size_t headElems = (size_t)BB * HH * SS * DHH;   // 4M halves = 8 MB
  h16* Qh  = (h16*)d_ws;
  h16* Kh  = Qh + headElems;
  h16* Vt  = Kh + headElems;
  h16* ctx = Vt + headElems;

  const dim3 gridGemm(DD / 128, (BB * SS) / 128);        // 8 x 32
  const float inv_sqrt_dh = 0.125f;                      // 1/sqrt(64), folded into Q

  gemm_nt_kernel<0, false><<<gridGemm, 256, 0, stream>>>(q, Wq, bq, Qh, inv_sqrt_dh);
  gemm_nt_kernel<0, false><<<gridGemm, 256, 0, stream>>>(k, Wk, bk, Kh, 1.0f);
  gemm_nt_kernel<1, false><<<gridGemm, 256, 0, stream>>>(v, Wv, bv, Vt, 1.0f);

  attn_scores_kernel<<<BB * HH * (SS / 16), 256, 16 * SS * sizeof(float), stream>>>(
      Qh, Kh, attn);

  attn_pv_kernel<<<BB * HH * (SS / 128), 256, 0, stream>>>(attn, Vt, ctx);

  gemm_nt_kernel<2, true><<<gridGemm, 256, 0, stream>>>(ctx, Wo, bo, out, 1.0f);
}
